// GlobalMHSA_18571438588113
// MI455X (gfx1250) — matrix operand order repaired and checked by reading, not compiled, not run on hardware
//
#include <hip/hip_runtime.h>
#include <hip/hip_bf16.h>

// ---------------------------------------------------------------------------
// GlobalMHSA on gfx1250 (MI455X): LN -> QKV WMMA GEMM -> flash attention with
// rel-pos bias (WMMA bf16, online softmax, async-LDS K/V staging) ->
// projection WMMA GEMM + residual.  All matmuls: v_wmma_f32_16x16x32_bf16.
// GEMM waves compute 32x64 tiles (8 WMMAs per 32-deep k-step) with
// double-buffered operands for load/math overlap.
// ---------------------------------------------------------------------------

typedef __attribute__((ext_vector_type(16))) __bf16 bf16x16;
typedef __attribute__((ext_vector_type(8)))  float  f32x8;

union BV { bf16x16 v; uint4 q[2]; };

#define WMMA_BF16(A, B, C) \
  __builtin_amdgcn_wmma_f32_16x16x32_bf16(false, (A), false, (B), (short)0, (C), false, false)

static constexpr int DIM      = 512;
static constexpr int HEADS    = 8;
static constexpr int DHEAD    = 64;
static constexpr int BATCH    = 2;
static constexpr int SEQ      = 2048;
static constexpr int ROWS     = BATCH * SEQ;       // 4096
static constexpr int QKV_N    = 3 * DIM;           // 1536
static constexpr int TBL_MAX  = 2 * 4096 - 2;      // 8190 (clamp upper bound)

// workspace layout (bytes)
static constexpr size_t XN_OFF    = 0;                                  // [4096,512] bf16
static constexpr size_t WQKVT_OFF = XN_OFF    + (size_t)ROWS * DIM * 2; // [1536,512] bf16
static constexpr size_t WPROJT_OFF= WQKVT_OFF + (size_t)QKV_N * DIM * 2;// [512,512]  bf16
static constexpr size_t Q_OFF     = WPROJT_OFF+ (size_t)DIM * DIM * 2;  // [B,H,T,64] bf16
static constexpr size_t K_OFF     = Q_OFF     + (size_t)ROWS * DIM * 2; // [B,H,T,64] bf16
static constexpr size_t VT_OFF    = K_OFF     + (size_t)ROWS * DIM * 2; // [B,H,64,T] bf16
static constexpr size_t AO_OFF    = VT_OFF    + (size_t)ROWS * DIM * 2; // [4096,512] bf16

// Per-lane async copy: 16 bytes global -> LDS (ASYNCcnt-tracked, CDNA5 path).
__device__ __forceinline__ void async_copy_b128(unsigned lds_byte, const void* gaddr) {
  asm volatile("global_load_async_to_lds_b128 %0, %1, off"
               :: "v"(lds_byte), "v"((unsigned long long)(uintptr_t)gaddr)
               : "memory");
}
__device__ __forceinline__ unsigned lds_off32(const void* p) {
  // generic pointer to LDS: low 32 bits are the LDS byte offset
  return (unsigned)(uintptr_t)p;
}

// ---------------------------------------------------------------------------
// 1) Transpose + bf16-convert weights.
// ---------------------------------------------------------------------------
__global__ void prep_weights_kernel(const float* __restrict__ wqkv,
                                    const float* __restrict__ wproj,
                                    __bf16* __restrict__ wqkvT,
                                    __bf16* __restrict__ wprojT) {
  int idx = blockIdx.x * 256 + threadIdx.x;
  if (idx < QKV_N * DIM) {
    int n = idx / DIM, k = idx % DIM;
    wqkvT[idx] = (__bf16)wqkv[k * QKV_N + n];
  } else {
    int i = idx - QKV_N * DIM;
    int n = i / DIM, k = i % DIM;
    wprojT[i] = (__bf16)wproj[k * DIM + n];
  }
}

// ---------------------------------------------------------------------------
// 2) LayerNorm: one wave per row of 512; bf16 output.
// ---------------------------------------------------------------------------
__global__ void layernorm_kernel(const float* __restrict__ x,
                                 const float* __restrict__ nw,
                                 const float* __restrict__ nb,
                                 __bf16* __restrict__ xn) {
  int wave = threadIdx.x >> 5, lane = threadIdx.x & 31;
  int row  = blockIdx.x * 8 + wave;
  const float* xr = x + (size_t)row * DIM;
  float4 v[4];
  float s = 0.f, ss = 0.f;
#pragma unroll
  for (int i = 0; i < 4; i++) {
    v[i] = *(const float4*)(xr + lane * 16 + i * 4);
    s  += v[i].x + v[i].y + v[i].z + v[i].w;
    ss += v[i].x * v[i].x + v[i].y * v[i].y + v[i].z * v[i].z + v[i].w * v[i].w;
  }
#pragma unroll
  for (int off = 1; off < 32; off <<= 1) {
    s  += __shfl_xor(s,  off, 32);
    ss += __shfl_xor(ss, off, 32);
  }
  float mu   = s * (1.f / DIM);
  float var  = ss * (1.f / DIM) - mu * mu;
  float rinv = rsqrtf(var + 1e-5f);
  __bf16* xo = xn + (size_t)row * DIM;
#pragma unroll
  for (int i = 0; i < 4; i++) {
    int c0 = lane * 16 + i * 4;
    float e[4] = {v[i].x, v[i].y, v[i].z, v[i].w};
#pragma unroll
    for (int j = 0; j < 4; j++)
      xo[c0 + j] = (__bf16)(((e[j] - mu) * rinv) * nw[c0 + j] + nb[c0 + j]);
  }
}

// ---------------------------------------------------------------------------
// Shared GEMM core: 32x64 wave tile, double-buffered k-loop.
// 8 WMMAs per k-step; each B tile feeds 2 WMMAs, each A tile feeds 4.
// ---------------------------------------------------------------------------
__device__ __forceinline__ void gemm_wave_32x64(const __bf16* __restrict__ abase0,
                                                const __bf16* __restrict__ abase1,
                                                const __bf16* const bbase[4],
                                                int hf, f32x8 acc[2][4]) {
  BV a[2][2], bm[2][4];

  auto loadA = [&](int buf, int kk) {
    a[buf][0].q[0] = *(const uint4*)(abase0 + kk + 16 * hf);
    a[buf][0].q[1] = *(const uint4*)(abase0 + kk + 16 * hf + 8);
    a[buf][1].q[0] = *(const uint4*)(abase1 + kk + 16 * hf);
    a[buf][1].q[1] = *(const uint4*)(abase1 + kk + 16 * hf + 8);
  };
  auto loadB = [&](int buf, int kk) {
#pragma unroll
    for (int c = 0; c < 4; c++) {
      bm[buf][c].q[0] = *(const uint4*)(bbase[c] + kk);
      bm[buf][c].q[1] = *(const uint4*)(bbase[c] + kk + 8);
    }
  };

  loadA(0, 0);
  loadB(0, 0);
#pragma unroll
  for (int i = 0; i < DIM / 32; i++) {
    int cur = i & 1, nxt = cur ^ 1;
    int kn = ((i + 1) & (DIM / 32 - 1)) * 32;   // wrap keeps flow uniform
    loadA(nxt, kn);
    loadB(nxt, kn);
#pragma unroll
    for (int c = 0; c < 4; c++) {
      acc[0][c] = WMMA_BF16(a[cur][0].v, bm[cur][c].v, acc[0][c]);
      acc[1][c] = WMMA_BF16(a[cur][1].v, bm[cur][c].v, acc[1][c]);
    }
  }
}

// ---------------------------------------------------------------------------
// 3) QKV GEMM: xn[4096,512] x wqkvT -> Q[B,H,T,64], K[B,H,T,64], V^T[B,H,64,T]
//    Wave tile 32x64; block = 8 waves = 64x256.
// ---------------------------------------------------------------------------
__global__ void qkv_gemm_kernel(const __bf16* __restrict__ xn,
                                const __bf16* __restrict__ wT,
                                __bf16* __restrict__ Qb,
                                __bf16* __restrict__ Kb,
                                __bf16* __restrict__ VTb) {
  int lane = threadIdx.x & 31, wave = threadIdx.x >> 5;
  int hf = lane >> 4, l16 = lane & 15;
  int m0 = blockIdx.x * 64 + (wave & 1) * 32;
  int n0 = blockIdx.y * 256 + (wave >> 1) * 64;

  const __bf16* abase0 = xn + (size_t)(m0 + l16) * DIM;
  const __bf16* abase1 = xn + (size_t)(m0 + 16 + l16) * DIM;
  const __bf16* bbase[4];
#pragma unroll
  for (int c = 0; c < 4; c++)
    bbase[c] = wT + (size_t)(n0 + 16 * c + l16) * DIM + 16 * hf;

  f32x8 acc[2][4] = {};
  gemm_wave_32x64(abase0, abase1, bbase, hf, acc);

#pragma unroll
  for (int c = 0; c < 4; c++) {
    int j     = n0 + 16 * c + l16;       // 0..1535
    int which = j >> 9;                  // 0=Q 1=K 2=V
    int hh    = (j >> 6) & 7;
    int d     = j & 63;
#pragma unroll
    for (int mi = 0; mi < 2; mi++) {
#pragma unroll
      for (int r = 0; r < 8; r++) {
        int row = m0 + 16 * mi + r + 8 * hf;
        int b = row >> 11, t = row & (SEQ - 1);
        int bh = b * HEADS + hh;
        __bf16 val = (__bf16)acc[mi][c][r];
        if (which == 0)      Qb[((size_t)bh * SEQ + t) * DHEAD + d] = val;
        else if (which == 1) Kb[((size_t)bh * SEQ + t) * DHEAD + d] = val;
        else                 VTb[((size_t)bh * DHEAD + d) * SEQ + t] = val;
      }
    }
  }
}

// ---------------------------------------------------------------------------
// 4) Flash attention.  One wave per 16-query tile; the 8 waves of a block
//    share each 32-key K/V block, staged into LDS with double-buffered
//    global_load_async_to_lds_b128 (ASYNCcnt) so the copy for block i+1
//    overlaps compute on block i.  S/P never touch HBM.
// ---------------------------------------------------------------------------
__global__ void attn_kernel(const __bf16* __restrict__ Qb,
                            const __bf16* __restrict__ Kb,
                            const __bf16* __restrict__ VTb,
                            const float* __restrict__ bias_table,
                            __bf16* __restrict__ AO) {
  __shared__ __align__(16) __bf16 Kt[2][32 * DHEAD];   // [key][d]   4KB x2
  __shared__ __align__(16) __bf16 Vt[2][DHEAD * 32];   // [d][key]   4KB x2
  __shared__ __align__(16) __bf16 Pt[8][16 * 32];      // per-wave P staging

  int tid  = threadIdx.x;
  int lane = tid & 31, wave = tid >> 5;
  int hf = lane >> 4, l16 = lane & 15;
  int bh = blockIdx.y;                 // 0..15
  int hh = bh & (HEADS - 1);
  int b  = bh >> 3;
  int qb = blockIdx.x * 128 + wave * 16;

  const __bf16* Qh = Qb  + (size_t)bh * SEQ * DHEAD;
  const __bf16* Kh = Kb  + (size_t)bh * SEQ * DHEAD;
  const __bf16* Vh = VTb + (size_t)bh * DHEAD * SEQ;
  __bf16* lw = Pt[wave];

  // async copy addresses for this thread (16B per buffer per block)
  unsigned kt_l[2], vt_l[2];
  kt_l[0] = lds_off32(&Kt[0][0]) + tid * 16;
  kt_l[1] = lds_off32(&Kt[1][0]) + tid * 16;
  unsigned voff = ((unsigned)(tid >> 2) * 32 + (tid & 3) * 8) * 2;
  vt_l[0] = lds_off32(&Vt[0][0]) + voff;
  vt_l[1] = lds_off32(&Vt[1][0]) + voff;

  // Q tile as two 16x32 A-operands (Dh = 64)
  BV aq[2];
  {
    const __bf16* qp = Qh + (size_t)(qb + l16) * DHEAD;
#pragma unroll
    for (int c = 0; c < 2; c++) {
      aq[c].q[0] = *(const uint4*)(qp + 32 * c + 16 * hf);
      aq[c].q[1] = *(const uint4*)(qp + 32 * c + 16 * hf + 8);
    }
  }

  f32x8 O[4] = {};
  float mrow[8], lrow[8];
#pragma unroll
  for (int r = 0; r < 8; r++) { mrow[r] = -1e30f; lrow[r] = 0.f; }

  const float scale = 0.125f;  // 1/sqrt(64)

  // prime buffer 0 with key block 0
  async_copy_b128(kt_l[0], Kh + tid * 8);
  async_copy_b128(vt_l[0], Vh + (size_t)(tid >> 2) * SEQ + (tid & 3) * 8);

  for (int kb = 0; kb < SEQ; kb += 32) {
    int cur = (kb >> 5) & 1, nxt = cur ^ 1;
    int kn  = (kb + 32) & (SEQ - 1);     // wrap: uniform flow, harmless extra
    // prefetch next key block into the other buffer
    async_copy_b128(kt_l[nxt], Kh + (size_t)kn * DHEAD + tid * 8);
    async_copy_b128(vt_l[nxt], Vh + (size_t)(tid >> 2) * SEQ + kn + (tid & 3) * 8);
    // wait for current buffer only (<=2 outstanding = the two just issued)
    asm volatile("s_wait_asynccnt 0x2" ::: "memory");
    __syncthreads();

    const __bf16* Kl = Kt[cur];
    const __bf16* Vl = Vt[cur];

    // ---- S = Q K^T : two 16x16 tiles, B-operands from LDS (ds_load_b128)
    f32x8 S[2] = {};
#pragma unroll
    for (int n = 0; n < 2; n++) {
      const __bf16* kp = Kl + (16 * n + l16) * DHEAD + 16 * hf;
      BV b0, b1;
      b0.q[0] = *(const uint4*)(kp);        b0.q[1] = *(const uint4*)(kp + 8);
      b1.q[0] = *(const uint4*)(kp + 32);   b1.q[1] = *(const uint4*)(kp + 40);
      S[n] = WMMA_BF16(aq[0].v, b0.v, S[n]);
      S[n] = WMMA_BF16(aq[1].v, b1.v, S[n]);
    }

    // ---- scale + rel-pos bias; row-wise online softmax
    float p0[8], p1[8], rm[8];
#pragma unroll
    for (int r = 0; r < 8; r++) {
      int q  = qb + r + 8 * hf;
      int k0 = kb + l16, k1 = kb + 16 + l16;
      int i0 = q - k0 + SEQ - 1;  i0 = i0 < 0 ? 0 : (i0 > TBL_MAX ? TBL_MAX : i0);
      int i1 = q - k1 + SEQ - 1;  i1 = i1 < 0 ? 0 : (i1 > TBL_MAX ? TBL_MAX : i1);
      p0[r] = S[0][r] * scale + bias_table[i0 * HEADS + hh];
      p1[r] = S[1][r] * scale + bias_table[i1 * HEADS + hh];
      rm[r] = fmaxf(p0[r], p1[r]);
    }
#pragma unroll
    for (int off = 1; off < 16; off <<= 1)
#pragma unroll
      for (int r = 0; r < 8; r++) rm[r] = fmaxf(rm[r], __shfl_xor(rm[r], off, 32));
    float alpha[8], rs[8];
#pragma unroll
    for (int r = 0; r < 8; r++) {
      float mn = fmaxf(mrow[r], rm[r]);
      alpha[r] = __expf(mrow[r] - mn);
      mrow[r]  = mn;
      p0[r] = __expf(p0[r] - mn);
      p1[r] = __expf(p1[r] - mn);
      rs[r] = p0[r] + p1[r];
    }
#pragma unroll
    for (int off = 1; off < 16; off <<= 1)
#pragma unroll
      for (int r = 0; r < 8; r++) rs[r] += __shfl_xor(rs[r], off, 32);
#pragma unroll
    for (int r = 0; r < 8; r++) lrow[r] = lrow[r] * alpha[r] + rs[r];
#pragma unroll
    for (int c = 0; c < 4; c++)
#pragma unroll
      for (int r = 0; r < 8; r++) O[c][r] *= alpha[r];

    // ---- P: C-layout -> A-layout via wave-private LDS slice
#pragma unroll
    for (int r = 0; r < 8; r++) {
      lw[(r + 8 * hf) * 32 + l16]      = (__bf16)p0[r];
      lw[(r + 8 * hf) * 32 + 16 + l16] = (__bf16)p1[r];
    }
    asm volatile("s_wait_dscnt 0x0" ::: "memory");   // wave-local DS ordering
    BV apm;
    apm.q[0] = *(const uint4*)(lw + l16 * 32 + 16 * hf);
    apm.q[1] = *(const uint4*)(lw + l16 * 32 + 16 * hf + 8);

    // ---- O += P V   (B-operand from LDS V^T tile)
#pragma unroll
    for (int c = 0; c < 4; c++) {
      const __bf16* vp = Vl + (16 * c + l16) * 32 + 16 * hf;
      BV bv;
      bv.q[0] = *(const uint4*)(vp);
      bv.q[1] = *(const uint4*)(vp + 8);
      O[c] = WMMA_BF16(apm.v, bv.v, O[c]);
    }
    // protect current buffer from next iteration's prefetch overwrite
    __syncthreads();
  }

  // ---- normalize and emit attn output [4096, 512] bf16 (heads gathered)
#pragma unroll
  for (int c = 0; c < 4; c++) {
    int d = 16 * c + l16;
#pragma unroll
    for (int r = 0; r < 8; r++) {
      int q = qb + r + 8 * hf;
      AO[((size_t)(b * SEQ + q)) * DIM + hh * DHEAD + d] =
          (__bf16)(O[c][r] / lrow[r]);
    }
  }
}

// ---------------------------------------------------------------------------
// 5) Projection GEMM + bias + residual: AO[4096,512] x wprojT + x -> out f32
//    Wave tile 32x64; block = 8 waves = 64x256.
// ---------------------------------------------------------------------------
__global__ void proj_gemm_kernel(const __bf16* __restrict__ AO,
                                 const __bf16* __restrict__ wT,
                                 const float* __restrict__ x,
                                 const float* __restrict__ bproj,
                                 float* __restrict__ out) {
  int lane = threadIdx.x & 31, wave = threadIdx.x >> 5;
  int hf = lane >> 4, l16 = lane & 15;
  int m0 = blockIdx.x * 64 + (wave & 1) * 32;
  int n0 = blockIdx.y * 256 + (wave >> 1) * 64;

  const __bf16* abase0 = AO + (size_t)(m0 + l16) * DIM;
  const __bf16* abase1 = AO + (size_t)(m0 + 16 + l16) * DIM;
  const __bf16* bbase[4];
#pragma unroll
  for (int c = 0; c < 4; c++)
    bbase[c] = wT + (size_t)(n0 + 16 * c + l16) * DIM + 16 * hf;

  f32x8 acc[2][4] = {};
  gemm_wave_32x64(abase0, abase1, bbase, hf, acc);

#pragma unroll
  for (int c = 0; c < 4; c++) {
    int col = n0 + 16 * c + l16;
#pragma unroll
    for (int mi = 0; mi < 2; mi++) {
#pragma unroll
      for (int r = 0; r < 8; r++) {
        int row = m0 + 16 * mi + r + 8 * hf;
        size_t o = (size_t)row * DIM + col;
        out[o] = x[o] + acc[mi][c][r] + bproj[col];
      }
    }
  }
}

// ---------------------------------------------------------------------------
extern "C" void kernel_launch(void* const* d_in, const int* in_sizes, int n_in,
                              void* d_out, int out_size, void* d_ws, size_t ws_size,
                              hipStream_t stream) {
  const float* x          = (const float*)d_in[0];
  const float* norm_w     = (const float*)d_in[1];
  const float* norm_b     = (const float*)d_in[2];
  const float* w_qkv      = (const float*)d_in[3];
  const float* w_proj     = (const float*)d_in[4];
  const float* b_proj     = (const float*)d_in[5];
  const float* bias_table = (const float*)d_in[6];
  float* out = (float*)d_out;

  char* ws = (char*)d_ws;
  __bf16* xn     = (__bf16*)(ws + XN_OFF);
  __bf16* wqkvT  = (__bf16*)(ws + WQKVT_OFF);
  __bf16* wprojT = (__bf16*)(ws + WPROJT_OFF);
  __bf16* Qb     = (__bf16*)(ws + Q_OFF);
  __bf16* Kb     = (__bf16*)(ws + K_OFF);
  __bf16* VTb    = (__bf16*)(ws + VT_OFF);
  __bf16* AO     = (__bf16*)(ws + AO_OFF);

  prep_weights_kernel<<<(QKV_N * DIM + DIM * DIM) / 256, 256, 0, stream>>>(
      w_qkv, w_proj, wqkvT, wprojT);
  layernorm_kernel<<<ROWS / 8, 256, 0, stream>>>(x, norm_w, norm_b, xn);
  qkv_gemm_kernel<<<dim3(ROWS / 64, QKV_N / 256), 256, 0, stream>>>(
      xn, wqkvT, Qb, Kb, VTb);
  attn_kernel<<<dim3(SEQ / 128, BATCH * HEADS), 256, 0, stream>>>(
      Qb, Kb, VTb, bias_table, AO);
  proj_gemm_kernel<<<dim3(ROWS / 64, DIM / 256), 256, 0, stream>>>(
      AO, wprojT, x, b_proj, out);
}
